// MyVIT_46780783788118
// MI455X (gfx1250) — compile-verified, matrix operand order checked by reading, not run on hardware
//
#include <hip/hip_runtime.h>
#include <hip/hip_bf16.h>
#include <stdint.h>

// ---------------------------------------------------------------------------
// ViT patch embedding on MI455X (gfx1250, wave32, WMMA + TDM + async-LDS).
//   M = 256*64 = 16384 patch rows, K = 2352 (padded to 2368), N = 512.
//   fp32 inputs -> bf16 operands, fp32 accumulation via v_wmma_f32_16x16x32_bf16.
// ---------------------------------------------------------------------------

#define BATCH   256
#define NPATCH  64
#define IN_D    2352
#define KP      2368            // IN_D padded to multiple of 32 (74 k-steps)
#define HID     512
#define SEQ     65
#define MTOT    (BATCH * NPATCH)    // 16384
#define NKSTEP  (KP / 32)           // 74
#define LDS_ROW 40                  // 32 bf16 + 8 pad -> conflict-free b128 reads
#define ATILE_E (128 * LDS_ROW)     // 5120 elements per tile buffer
#define BUF_E   (2 * ATILE_E)       // A + B region per double-buffer slot

typedef __attribute__((ext_vector_type(16))) __bf16 v16bf;
typedef __attribute__((ext_vector_type(8)))  float  v8f;
typedef __attribute__((ext_vector_type(4)))  unsigned v4u;
typedef __attribute__((ext_vector_type(8)))  int      v8i;
typedef __attribute__((ext_vector_type(4)))  int      v4i;

union FragBF { v16bf v; uint4 q[2]; };

// ---------------------------------------------------------------------------
// Repack/patchify X[B,3,224,224] -> A_bf16[MTOT][KP] (row = b*64 + py*8 + px,
// col k = c*784 + i*28 + j), zero padding for k in [2352,2368).
// ---------------------------------------------------------------------------
__global__ void vit_repack_a(const float* __restrict__ X,
                             __hip_bfloat16* __restrict__ A) {
    unsigned idx = blockIdx.x * 256u + threadIdx.x;
    if (idx >= (unsigned)MTOT * KP) return;
    unsigned k  = idx % KP;
    unsigned mp = idx / KP;
    float x = 0.0f;
    if (k < IN_D) {
        unsigned b  = mp >> 6, p = mp & 63u;
        unsigned py = p >> 3, px = p & 7u;
        unsigned c  = k / 784u, rem = k % 784u;
        unsigned i  = rem / 28u, j = rem % 28u;
        x = X[(((size_t)b * 3u + c) * 224u + (py * 28u + i)) * 224u + (px * 28u + j)];
    }
    A[idx] = __float2bfloat16(x);
}

// W[512][2352] fp32 -> Wb[512][2368] bf16 (zero-padded K tail)
__global__ void vit_repack_w(const float* __restrict__ W,
                             __hip_bfloat16* __restrict__ Wb) {
    unsigned idx = blockIdx.x * 256u + threadIdx.x;
    if (idx >= (unsigned)HID * KP) return;
    unsigned k = idx % KP;
    unsigned h = idx / KP;
    float x = (k < IN_D) ? W[(size_t)h * IN_D + k] : 0.0f;
    Wb[idx] = __float2bfloat16(x);
}

// out[b, 0, h] = class_token[h] * pos_embed[0, h]
__global__ void vit_cls_row(const float* __restrict__ ct,
                            const float* __restrict__ pos,
                            float* __restrict__ out) {
    unsigned i = blockIdx.x * 256u + threadIdx.x;
    if (i >= (unsigned)BATCH * HID) return;
    unsigned b = i >> 9, h = i & 511u;
    out[(size_t)b * (SEQ * HID) + h] = ct[h] * pos[h];
}

// ---------------------------------------------------------------------------
// Tiled GEMM: 128x128 tile per 256-thread block (8 waves), K-step 32.
// A tile: Tensor Data Mover (one D# per tile, HW inserts the LDS row padding).
// B tile: per-thread global_load_async_to_lds_b128.
// Double-buffered LDS, 8 WMMAs/wave/k-step.
// ---------------------------------------------------------------------------
__global__ __launch_bounds__(256) void vit_gemm(
    const __hip_bfloat16* __restrict__ A,    // [MTOT][KP]
    const __hip_bfloat16* __restrict__ Wb,   // [HID][KP]
    const float* __restrict__ bias,          // [HID]
    const float* __restrict__ pos,           // [65][512]
    float* __restrict__ out)                 // [B][65][512]
{
    __shared__ __align__(16) unsigned short smem[2 * BUF_E];  // 40 KB

    const unsigned tid  = threadIdx.x;
    const unsigned lane = tid & 31u, wid = tid >> 5;
    const unsigned m0 = blockIdx.y * 128u;
    const unsigned n0 = blockIdx.x * 128u;

    // ---- TDM issue: one 2D descriptor moves a 128x32-bf16 tile into LDS,
    //      padding each 64-byte row (16 DWORDs) with 16 bytes (4 DWORDs)
    //      to produce the LDS_ROW=40-element row stride.
    auto issue_tdm = [&](const __hip_bfloat16* gtile, unsigned lbase_elem) {
        unsigned long long ga = (unsigned long long)(uintptr_t)gtile;
        unsigned lds_byte = (unsigned)(uintptr_t)&smem[lbase_elem];
        v4u g0 = { 1u,                                       // count=1, user D#
                   lds_byte,                                 // lds_addr
                   (unsigned)ga,                             // global_addr[31:0]
                   (unsigned)((ga >> 32) & 0x01ffffffu) | (2u << 30) }; // [56:32] | type=2
        v8i g1 = { (int)((1u << 16)      // data_size = 2 bytes
                       | (1u << 20)      // pad_enable
                       | (3u << 22)      // pad_interval: 16 DWORDs
                       | (3u << 25)),    // pad_amount:   4 DWORDs
                   (int)((KP & 0xffffu) << 16),              // tensor_dim0 lo16
                   (int)(((unsigned)KP >> 16) | (128u << 16)),// dim0 hi16 | tensor_dim1 lo16
                   (int)(0u | (32u << 16)),                  // dim1 hi16 | tile_dim0=32
                   (int)(128u),                              // tile_dim1=128 | tile_dim2=0
                   (int)KP,                                  // tensor_dim0_stride lo32
                   0, 0 };                                   // stride hi | dim1_stride
        v4i g2 = { 0, 0, 0, 0 };
        v4i g3 = { 0, 0, 0, 0 };
        v8i g4 = { 0, 0, 0, 0, 0, 0, 0, 0 };
        __builtin_amdgcn_tensor_load_to_lds(g0, g1, g2, g3, g4, 0);
    };

    // ---- async-LDS issue: one 128x32-bf16 tile as 512 16B transfers.
    auto issue_async = [&](const __hip_bfloat16* gk, unsigned grow0, unsigned lbase) {
#pragma unroll
        for (int r2 = 0; r2 < 2; ++r2) {
            unsigned idx = tid + (unsigned)r2 * 256u;
            unsigned row = idx >> 2, seg = idx & 3u;
            unsigned voff  = ((grow0 + row) * (unsigned)KP + seg * 8u) * 2u; // bytes
            unsigned laddr = (unsigned)(uintptr_t)&smem[lbase + row * LDS_ROW + seg * 8u];
            unsigned long long sbase = (unsigned long long)(uintptr_t)gk;
            asm volatile("global_load_async_to_lds_b128 %0, %1, %2"
                         :: "v"(laddr), "v"(voff), "s"(sbase) : "memory");
        }
    };

    // Prologue: fill buffer 0 (A region at 0 via TDM, B region via async).
    if (wid == 0) issue_tdm(A + (size_t)m0 * KP, 0u);
    issue_async(Wb, n0, ATILE_E);
    asm volatile("s_wait_asynccnt 0" ::: "memory");
    if (wid == 0) __builtin_amdgcn_s_wait_tensorcnt(0);
    __syncthreads();

    // Wave sub-tile: 4 waves along M (32 rows each) x 2 waves along N (64 cols).
    const unsigned arow  = (wid >> 1) * 32u;
    const unsigned brow  = (wid & 1u) * 64u;
    const unsigned lcol  = lane & 15u;
    const unsigned lhalf = lane >> 4;     // K-half select per ISA A/B layout

    v8f acc[2][4] = {};

    for (int ks = 0; ks < NKSTEP; ++ks) {
        const unsigned curb = (unsigned)(ks & 1) * BUF_E;
        if (ks + 1 < NKSTEP) {
            const unsigned nxtb = (curb ^ BUF_E);
            const int kb = (ks + 1) * 32;
            if (wid == 0) issue_tdm(A + (size_t)m0 * KP + kb, nxtb);
            issue_async(Wb + kb, n0, nxtb + ATILE_E);
        }

        FragBF af[2], bf[4];
#pragma unroll
        for (int tm = 0; tm < 2; ++tm) {
            unsigned r    = arow + (unsigned)tm * 16u + lcol;
            unsigned base = curb + r * LDS_ROW + lhalf * 8u;
            af[tm].q[0] = *(const uint4*)&smem[base];        // K 0..7 / 8..15
            af[tm].q[1] = *(const uint4*)&smem[base + 16u];  // K 16..23 / 24..31
        }
#pragma unroll
        for (int tn = 0; tn < 4; ++tn) {
            unsigned r    = brow + (unsigned)tn * 16u + lcol;
            unsigned base = curb + ATILE_E + r * LDS_ROW + lhalf * 8u;
            bf[tn].q[0] = *(const uint4*)&smem[base];
            bf[tn].q[1] = *(const uint4*)&smem[base + 16u];
        }

#pragma unroll
        for (int tm = 0; tm < 2; ++tm)
#pragma unroll
            for (int tn = 0; tn < 4; ++tn)
                acc[tm][tn] = __builtin_amdgcn_wmma_f32_16x16x32_bf16(
                    false, af[tm].v, false, bf[tn].v,
                    (short)0, acc[tm][tn], false, false);

        asm volatile("s_wait_asynccnt 0" ::: "memory");
        if (wid == 0) __builtin_amdgcn_s_wait_tensorcnt(0);
        __syncthreads();
    }

    // Epilogue: D layout = 8 VGPRs; lane 0-15 -> M=r, lane 16-31 -> M=r+8; N=lane&15.
#pragma unroll
    for (int tm = 0; tm < 2; ++tm) {
#pragma unroll
        for (int tn = 0; tn < 4; ++tn) {
            unsigned n  = n0 + brow + (unsigned)tn * 16u + lcol;
            float    bv = bias[n];
            v8f c = acc[tm][tn];
#pragma unroll
            for (int r = 0; r < 8; ++r) {
                unsigned m = m0 + arow + (unsigned)tm * 16u + lhalf * 8u + (unsigned)r;
                unsigned b = m >> 6, p = m & 63u;
                float v = (c[r] + bv) * pos[(1u + p) * HID + n];
                out[((size_t)b * SEQ + 1u + p) * HID + n] = v;
            }
        }
    }
}

extern "C" void kernel_launch(void* const* d_in, const int* in_sizes, int n_in,
                              void* d_out, int out_size, void* d_ws, size_t ws_size,
                              hipStream_t stream) {
    const float* X  = (const float*)d_in[0];
    const float* W  = (const float*)d_in[1];
    const float* b  = (const float*)d_in[2];
    const float* ct = (const float*)d_in[3];
    const float* pe = (const float*)d_in[4];
    float* out = (float*)d_out;

    __hip_bfloat16* Abf = (__hip_bfloat16*)d_ws;                 // 16384*2368*2 B
    __hip_bfloat16* Wbf = Abf + (size_t)MTOT * KP;               // 512*2368*2 B

    {
        unsigned total = (unsigned)HID * KP;
        vit_repack_w<<<(total + 255u) / 256u, 256, 0, stream>>>(W, Wbf);
    }
    {
        unsigned total = (unsigned)MTOT * KP;
        vit_repack_a<<<(total + 255u) / 256u, 256, 0, stream>>>(X, Abf);
    }
    {
        unsigned total = (unsigned)BATCH * HID;
        vit_cls_row<<<(total + 255u) / 256u, 256, 0, stream>>>(ct, pe, out);
    }
    dim3 grid(HID / 128, MTOT / 128);   // (4, 128) tiles
    vit_gemm<<<grid, 256, 0, stream>>>(Abf, Wbf, b, pe, out);
}
